// CausalMultiHeadAttention_8074538517044
// MI455X (gfx1250) — matrix operand order repaired and checked by reading, not compiled, not run on hardware
//
#include <hip/hip_runtime.h>

typedef __attribute__((ext_vector_type(16))) __bf16 bf16x16;
typedef __attribute__((ext_vector_type(8)))  __bf16 bf16x8;
typedef __attribute__((ext_vector_type(2)))  __bf16 bf16x2;
typedef __attribute__((ext_vector_type(8)))  float  v8f;
typedef unsigned int u32x4 __attribute__((ext_vector_type(4)));
typedef int          i32x4 __attribute__((ext_vector_type(4)));
typedef int          i32x8 __attribute__((ext_vector_type(8)));

#define DEV __device__ __forceinline__

constexpr int B_ = 4, L_ = 2048, E_ = 256, H_ = 8, D_ = 256;
constexpr size_t HEADELEMS = (size_t)L_ * D_;               // 2048*256
constexpr size_t MATELEMS  = (size_t)B_ * H_ * HEADELEMS;   // 16,777,216
constexpr float  SCALE     = 0.0625f;                       // 1/sqrt(256)

#if defined(__has_builtin)
#  if __has_builtin(__builtin_amdgcn_tensor_load_to_lds)
#    define HAVE_TDM 1
#  else
#    define HAVE_TDM 0
#  endif
#else
#  define HAVE_TDM 0
#endif

DEV v8f wmma_bf16(bf16x16 a, bf16x16 b, v8f c) {
  return __builtin_amdgcn_wmma_f32_16x16x32_bf16(false, a, false, b,
                                                 (short)0, c, false, false);
}

// ---- Fragment loaders (layouts per CDNA5 ISA §7.12.2, wave32) ----
DEV bf16x16 load_a_f32(const float* __restrict__ src, int row, int ld, int k0, int lane) {
  const float* p = src + (size_t)(row + (lane & 15)) * ld + k0 + ((lane & 16) ? 8 : 0);
  bf16x16 a;
#pragma unroll
  for (int i = 0; i < 8; ++i) a[i] = (__bf16)p[i];
#pragma unroll
  for (int i = 0; i < 8; ++i) a[8 + i] = (__bf16)p[16 + i];
  return a;
}
DEV bf16x16 load_a_bf16(const __bf16* __restrict__ src, int row, int ld, int k0, int lane) {
  const __bf16* p = src + (size_t)(row + (lane & 15)) * ld + k0 + ((lane & 16) ? 8 : 0);
  bf16x16 a;
#pragma unroll
  for (int i = 0; i < 8; ++i) a[i] = p[i];
#pragma unroll
  for (int i = 0; i < 8; ++i) a[8 + i] = p[16 + i];
  return a;
}
// B 32x16: lane holds column N=lane&15; lanes 0-15 K=0..7,16..23; 16-31 K=8..15,24..31.
// Source is N-major with K contiguous: B[k][n] = src[(n0+n)*ld + k0+k].
DEV bf16x16 load_b_f32(const float* __restrict__ src, int n0, int ld, int k0, int lane) {
  const float* p = src + (size_t)(n0 + (lane & 15)) * ld + k0 + ((lane & 16) ? 8 : 0);
  bf16x16 b;
#pragma unroll
  for (int i = 0; i < 8; ++i) { b[i] = (__bf16)p[i]; b[8 + i] = (__bf16)p[16 + i]; }
  return b;
}
DEV bf16x16 load_b_bf16(const __bf16* src, int n0, int ld, int k0, int lane) {
  const __bf16* p = src + (size_t)(n0 + (lane & 15)) * ld + k0 + ((lane & 16) ? 8 : 0);
  bf16x16 b;
#pragma unroll
  for (int i = 0; i < 8; ++i) { b[i] = p[i]; b[8 + i] = p[16 + i]; }
  return b;
}

DEV float rmax16(float v) {
#pragma unroll
  for (int m = 1; m < 16; m <<= 1) v = fmaxf(v, __shfl_xor(v, m, 32));
  return v;
}
DEV float rsum16(float v) {
#pragma unroll
  for (int m = 1; m < 16; m <<= 1) v += __shfl_xor(v, m, 32);
  return v;
}

// ---------------------------------------------------------------------------
// Tensor Data Mover: 2D bf16 tile (tile_d0 contiguous elems x tile_d1 rows,
// row stride = stride0 elems) global -> LDS, packed row-major in LDS.
// D# layout per CDNA5 ISA.
// ---------------------------------------------------------------------------
#if HAVE_TDM
DEV void tdm_load_2d(unsigned lds_off, const void* gptr,
                     unsigned tensor_d0, unsigned tensor_d1,
                     unsigned tile_d0, unsigned tile_d1, unsigned stride0) {
  unsigned long long ga = (unsigned long long)(uintptr_t)gptr;
  u32x4 g0;
  g0[0] = 1u;                                        // count=1 (valid), user mode
  g0[1] = lds_off;                                   // lds_addr (bytes)
  g0[2] = (unsigned)ga;                              // global_addr[31:0]
  g0[3] = (unsigned)((ga >> 32) & 0x01FFFFFFull)     // global_addr[56:32]
          | (2u << 30);                              // type = 2 ("image")
  i32x8 g1;
  g1[0] = (int)(1u << 16);                           // data_size = 1 -> 2 bytes
  g1[1] = (int)((tensor_d0 & 0xffffu) << 16);        // tensor_dim0[15:0]
  g1[2] = (int)((tensor_d0 >> 16) | ((tensor_d1 & 0xffffu) << 16));
  g1[3] = (int)((tensor_d1 >> 16) | ((tile_d0 & 0xffffu) << 16));
  g1[4] = (int)(tile_d1 & 0xffffu);                  // tile_dim1; tile_dim2 = 0
  g1[5] = (int)stride0;                              // tensor_dim0_stride[31:0]
  g1[6] = 0;
  g1[7] = 0;
  i32x4 gz = {0, 0, 0, 0};
#if __clang_major__ <= 22
  __builtin_amdgcn_tensor_load_to_lds(g0, g1, gz, gz, 0);
#else
  i32x8 gz8 = {0, 0, 0, 0, 0, 0, 0, 0};
  __builtin_amdgcn_tensor_load_to_lds(g0, g1, gz, gz, gz8, 0);
#endif
}
DEV unsigned lds_offset(const void* p) { return (unsigned)(uintptr_t)p; }
#endif

// ---------------------------------------------------------------------------
// K1: Q/K/V projection.  y = x @ W^T.  M=8192, N=2048, K=256.
// gridDim.y selects Wq/Wk/Wv.  Q,K stored (b,h,l,d); V stored (b,h,d,l).
// ---------------------------------------------------------------------------
__global__ __launch_bounds__(256) void qkv_kernel(
    const float* __restrict__ x, const float* __restrict__ Wq,
    const float* __restrict__ Wk, const float* __restrict__ Wv,
    __bf16* __restrict__ Q, __bf16* __restrict__ Km, __bf16* __restrict__ Vt) {
  const int which = blockIdx.y;
  const float* W = (which == 0) ? Wq : (which == 1) ? Wk : Wv;
  __bf16* Out = (which == 0) ? Q : (which == 1) ? Km : Vt;
  const int lane = threadIdx.x & 31, wave = threadIdx.x >> 5;
  const int tile = blockIdx.x * 8 + wave;
  const int tn = tile & 127, tm = tile >> 7;
  const int m0 = tm * 16, n0 = tn * 16;
  v8f acc = {};
#pragma unroll
  for (int k0 = 0; k0 < E_; k0 += 32) {
    bf16x16 a = load_a_f32(x, m0, E_, k0, lane);
    bf16x16 b = load_b_f32(W, n0, E_, k0, lane);
    acc = wmma_bf16(a, b, acc);
  }
  const int half = lane >> 4, ln = lane & 15;
  const int n = n0 + ln, h = n >> 8, d = n & 255;
#pragma unroll
  for (int r = 0; r < 8; ++r) {
    const int m = m0 + half * 8 + r;
    const int bb = m >> 11, l = m & (L_ - 1);
    const size_t bhh = (size_t)(bb * H_ + h);
    if (which == 2)
      Out[(bhh * D_ + d) * L_ + l] = (__bf16)acc[r];
    else
      Out[(bhh * L_ + l) * D_ + d] = (__bf16)acc[r];
  }
}

// ---------------------------------------------------------------------------
// K2: full-row softmax stats.  Block = one (b,h) x 128 query rows; each wave
// owns one 16-row query tile.  K tiles TDM-staged into a double-buffered LDS
// slab: K[jb+1] streams in (tensorcnt<=1) underneath jb's WMMAs.
// ---------------------------------------------------------------------------
__global__ __launch_bounds__(256) void stats_kernel(
    const __bf16* __restrict__ Q, const __bf16* __restrict__ Km,
    float* __restrict__ stats) {
  __shared__ __bf16 sK[2][32 * 256];   // 2 x 16 KB, sK[buf][j_local*256 + d]
  const int lane = threadIdx.x & 31, wave = threadIdx.x >> 5;
  const int qb = blockIdx.x & 15, bh = blockIdx.x >> 4;
  const int qt = qb * 8 + wave, m0 = qt * 16;
  const int half = lane >> 4, ln = lane & 15;
  const __bf16* Qb = Q + (size_t)bh * HEADELEMS;
  const __bf16* Kb = Km + (size_t)bh * HEADELEMS;
  bf16x16 qa[8];
#pragma unroll
  for (int c = 0; c < 8; ++c) qa[c] = load_a_bf16(Qb, m0, D_, c * 32, lane);
  float mrun[8], zrun[8];
#pragma unroll
  for (int r = 0; r < 8; ++r) { mrun[r] = -3.0e38f; zrun[r] = 0.f; }
  v8f zero = {};
  constexpr int NJB = L_ / 32;   // 64
#if HAVE_TDM
  if (wave == 0) tdm_load_2d(lds_offset(sK[0]), Kb, D_, L_, D_, 32, D_);
#endif
  for (int jb = 0; jb < NJB; ++jb) {
#if HAVE_TDM
    const int cur = jb & 1;
    if (wave == 0) {
      if (jb + 1 < NJB) {
        tdm_load_2d(lds_offset(sK[cur ^ 1]), Kb + (size_t)(jb + 1) * 32 * D_,
                    D_, L_, D_, 32, D_);
        __builtin_amdgcn_s_wait_tensorcnt(1);   // K[jb] done, K[jb+1] in flight
      } else {
        __builtin_amdgcn_s_wait_tensorcnt(0);
      }
    }
#else
    const int cur = 0;
    {
      const uint4* gs = (const uint4*)(Kb + (size_t)jb * 32 * D_);
      uint4* ls = (uint4*)sK[0];
      for (int i = threadIdx.x; i < 1024; i += 256) ls[i] = gs[i];
    }
#endif
    __syncthreads();
    const __bf16* kbuf = sK[cur];
    v8f s0 = zero, s1 = zero;
#pragma unroll
    for (int c = 0; c < 8; ++c) {
      bf16x16 b0 = load_b_bf16(kbuf, 0, D_, c * 32, lane);
      bf16x16 b1 = load_b_bf16(kbuf, 16, D_, c * 32, lane);
      s0 = wmma_bf16(qa[c], b0, s0);
      s1 = wmma_bf16(qa[c], b1, s1);
    }
#pragma unroll
    for (int r = 0; r < 8; ++r) {
      const float v0 = s0[r] * SCALE, v1 = s1[r] * SCALE;
      const float tmax = fmaxf(rmax16(v0), rmax16(v1));
      const float nm = fmaxf(mrun[r], tmax);
      const float e = rsum16(__expf(v0 - nm)) + rsum16(__expf(v1 - nm));
      zrun[r] = zrun[r] * __expf(mrun[r] - nm) + e;
      mrun[r] = nm;
    }
    __syncthreads();
  }
  if (ln == 0) {
#pragma unroll
    for (int r = 0; r < 8; ++r) {
      const size_t gr = (size_t)bh * L_ + m0 + half * 8 + r;
      stats[2 * gr] = mrun[r];
      stats[2 * gr + 1] = zrun[r];
    }
  }
}

// ---------------------------------------------------------------------------
// K3: causal weighted-V pass.  sK double-buffered (K[jb+1] prefetch fully
// hidden); sV single-buffered with a split tensorcnt wait so V[jb] streams in
// under the S WMMAs.  Prob tile stored as packed bf16 pairs (cvt_pk + b32 DS
// stores), read back as two 16B DS loads + one shuffle.
// ---------------------------------------------------------------------------
__global__ __launch_bounds__(256) void attn_kernel(
    const __bf16* __restrict__ Q, const __bf16* __restrict__ Km,
    const __bf16* __restrict__ Vt, const float* __restrict__ stats,
    __bf16* __restrict__ O) {
  __shared__ __bf16 sK[2][32 * 256];   // 32 KB
  __shared__ __bf16 sV[256 * 32];      // 16 KB, sV[d*32 + j_local]
  __shared__ bf16x2 sP[8][16 * 16];    // 8 KB: [wave][row*16 + n] = (col n, col n+16)
  const int lane = threadIdx.x & 31, wave = threadIdx.x >> 5;
  const int qb = blockIdx.x & 15, bh = blockIdx.x >> 4;
  const int qt = qb * 8 + wave, m0 = qt * 16;
  const int half = lane >> 4, ln = lane & 15;
  const __bf16* Qb = Q + (size_t)bh * HEADELEMS;
  const __bf16* Kb = Km + (size_t)bh * HEADELEMS;
  const __bf16* Vb = Vt + (size_t)bh * HEADELEMS;   // (d, l) layout

  bf16x16 qa[8];
#pragma unroll
  for (int c = 0; c < 8; ++c) qa[c] = load_a_bf16(Qb, m0, D_, c * 32, lane);

  float mi[8], iz[8];
#pragma unroll
  for (int r = 0; r < 8; ++r) {
    const size_t gr = (size_t)bh * L_ + m0 + half * 8 + r;
    mi[r] = stats[2 * gr];
    iz[r] = 1.0f / stats[2 * gr + 1];
  }

  v8f zero = {};
  v8f oacc[16];
#pragma unroll
  for (int t = 0; t < 16; ++t) oacc[t] = zero;
  float zacc[8];
#pragma unroll
  for (int r = 0; r < 8; ++r) zacc[r] = 0.f;

  const int jmaxw = (m0 + 15) >> 5;      // last causal key block for this wave
  const int jmaxb = qb * 4 + 3;          // last key block any wave needs

#if HAVE_TDM
  if (wave == 0) tdm_load_2d(lds_offset(sK[0]), Kb, D_, L_, D_, 32, D_);
#endif
  for (int jb = 0; jb <= jmaxb; ++jb) {
    const int j0 = jb * 32;
#if HAVE_TDM
    const int cur = jb & 1;
    if (wave == 0) {
      tdm_load_2d(lds_offset(sV), Vb + j0, L_, D_, 32, D_, L_);
      if (jb < jmaxb) {
        tdm_load_2d(lds_offset(sK[cur ^ 1]), Kb + (size_t)(j0 + 32) * D_,
                    D_, L_, D_, 32, D_);
        __builtin_amdgcn_s_wait_tensorcnt(2);   // K[jb] done
      } else {
        __builtin_amdgcn_s_wait_tensorcnt(1);   // K[jb] done
      }
    }
#else
    const int cur = 0;
    {
      const uint4* gk = (const uint4*)(Kb + (size_t)j0 * D_);
      uint4* lk = (uint4*)sK[0];
      for (int i = threadIdx.x; i < 1024; i += 256) lk[i] = gk[i];
      uint4* lv = (uint4*)sV;
      for (int i = threadIdx.x; i < 1024; i += 256) {
        const int row = i >> 2, c = i & 3;
        lv[i] = *(const uint4*)(Vb + (size_t)row * L_ + j0 + c * 8);
      }
    }
#endif
    __syncthreads();                       // sK[cur] visible to all waves
    const __bf16* kbuf = sK[cur];
    if (jb <= jmaxw) {
      v8f s0 = zero, s1 = zero;
#pragma unroll
      for (int c = 0; c < 8; ++c) {
        bf16x16 b0 = load_b_bf16(kbuf, 0, D_, c * 32, lane);
        bf16x16 b1 = load_b_bf16(kbuf, 16, D_, c * 32, lane);
        s0 = wmma_bf16(qa[c], b0, s0);
        s1 = wmma_bf16(qa[c], b1, s1);
      }
#pragma unroll
      for (int r = 0; r < 8; ++r) {
        const int ig = m0 + half * 8 + r;
        const float p0 = __expf(s0[r] * SCALE - mi[r]) * iz[r];
        const float p1 = __expf(s1[r] * SCALE - mi[r]) * iz[r];
        const float w0 = (j0 + ln <= ig) ? __expf(p0) : 0.f;
        const float w1 = (j0 + 16 + ln <= ig) ? __expf(p1) : 0.f;
        zacc[r] += w0 + w1;
        bf16x2 pk;
        pk[0] = (__bf16)w0;
        pk[1] = (__bf16)w1;
        sP[wave][(half * 8 + r) * 16 + ln] = pk;   // cvt_pk + ds_store_b32
      }
    }
#if HAVE_TDM
    if (wave == 0) {
      if (jb < jmaxb) __builtin_amdgcn_s_wait_tensorcnt(1);  // V[jb] done
      else            __builtin_amdgcn_s_wait_tensorcnt(0);
    }
    __syncthreads();                       // sV visible to all waves
#endif
    if (jb <= jmaxw) {
      __builtin_amdgcn_wave_barrier();     // order same-wave DS store->load
      // A-fragment of probs: pairs kb..kb+7 of row ln; de-interleave lo/hi.
      const __bf16* base = (const __bf16*)&sP[wave][ln * 16 + (half ? 8 : 0)];
      bf16x8 lo = *(const bf16x8*)base;
      bf16x8 hi = *(const bf16x8*)(base + 8);
      bf16x16 pa = __builtin_shufflevector(lo, hi,
          0, 2, 4, 6, 8, 10, 12, 14, 1, 3, 5, 7, 9, 11, 13, 15);
#pragma unroll
      for (int t = 0; t < 16; ++t) {
        bf16x16 vb = load_b_bf16(sV, t * 16, 32, 0, lane);
        oacc[t] = wmma_bf16(pa, vb, oacc[t]);
      }
    }
    __syncthreads();                       // protect sV / sK[cur^1] reuse
  }
  // normalize and store this wave's 16x256 output tile
  float zfin[8];
#pragma unroll
  for (int r = 0; r < 8; ++r) zfin[r] = 1.0f / rsum16(zacc[r]);
  __bf16* Og = O + ((size_t)bh * L_ + m0) * D_;
#pragma unroll
  for (int t = 0; t < 16; ++t)
#pragma unroll
    for (int r = 0; r < 8; ++r)
      Og[(half * 8 + r) * D_ + t * 16 + ln] = (__bf16)(oacc[t][r] * zfin[r]);
}

// ---------------------------------------------------------------------------
// K4: output projection.  out = O2 @ Wo^T;  M=8192, N=256, K=2048 (8 heads).
// ---------------------------------------------------------------------------
__global__ __launch_bounds__(256) void proj_o_kernel(
    const __bf16* __restrict__ O, const float* __restrict__ Wo,
    float* __restrict__ out) {
  const int lane = threadIdx.x & 31, wave = threadIdx.x >> 5;
  const int tile = blockIdx.x * 8 + wave;
  const int tn = tile & 15, tm = tile >> 4;
  const int m0 = tm * 16, n0 = tn * 16;
  const int bb = m0 >> 11, l0 = m0 & (L_ - 1);
  v8f acc = {};
  for (int h = 0; h < H_; ++h) {
    const __bf16* Ob = O + ((size_t)(bb * H_ + h) * L_ + l0) * D_;
#pragma unroll
    for (int c = 0; c < 8; ++c) {
      bf16x16 a = load_a_bf16(Ob, 0, D_, c * 32, lane);
      bf16x16 b = load_b_f32(Wo, n0, 2048, h * D_ + c * 32, lane);
      acc = wmma_bf16(a, b, acc);
    }
  }
  const int half = lane >> 4, ln = lane & 15;
  const int n = n0 + ln;
#pragma unroll
  for (int r = 0; r < 8; ++r) {
    const int m = m0 + half * 8 + r;
    out[(size_t)m * E_ + n] = acc[r];
  }
}

// ---------------------------------------------------------------------------
extern "C" void kernel_launch(void* const* d_in, const int* in_sizes, int n_in,
                              void* d_out, int out_size, void* d_ws, size_t ws_size,
                              hipStream_t stream) {
  const float* x  = (const float*)d_in[0];
  const float* Wq = (const float*)d_in[1];
  const float* Wk = (const float*)d_in[2];
  const float* Wv = (const float*)d_in[3];
  const float* Wo = (const float*)d_in[4];
  float* out = (float*)d_out;

  __bf16* Q    = (__bf16*)d_ws;
  __bf16* Km   = Q + MATELEMS;
  __bf16* Vt   = Km + MATELEMS;          // transposed, (b,h,d,l)
  __bf16* O    = Vt + MATELEMS;
  float*  stats = (float*)(O + MATELEMS);

  qkv_kernel<<<dim3(8192, 3, 1), 256, 0, stream>>>(x, Wq, Wk, Wv, Q, Km, Vt);
  stats_kernel<<<dim3(512, 1, 1), 256, 0, stream>>>(Q, Km, stats);
  attn_kernel<<<dim3(512, 1, 1), 256, 0, stream>>>(Q, Km, Vt, stats, O);
  proj_o_kernel<<<dim3(1024, 1, 1), 256, 0, stream>>>(O, Wo, out);
}
